// PsiSparseAttention_10934986736080
// MI455X (gfx1250) — matrix operand order repaired and checked by reading, not compiled, not run on hardware
//
#include <hip/hip_runtime.h>

// PsiSparseAttention for MI455X (gfx1250, wave32, WMMA).
// B=4 H=16 S=1024 D=64 fp32. Global quantile(0.5) threshold -> masked softmax -> PV.
// Pass 1: WMMA QK^T -> LDS histogram (ds_add) -> global bins.
// Pass 2: serial scan -> threshold (linear interpolation, rank 0.5*(N-1)).
// Pass 3: flash attention, K/V^T tiles double-buffered in LDS via
//         global_load_async_to_lds_* (ASYNCcnt), bf16 WMMA for QK^T and PV.

#define BH   64          // B*H
#define SEQ  1024
#define DH   64
#define NBINS 4096
#define HLO  (-8.0f)
#define HBW  (16.0f / (float)NBINS)   // 1/256
#define NEGMAX (-3.402823466e38f)
#define QK_SCALE 0.125f               // 1/sqrt(64)

typedef __attribute__((ext_vector_type(16))) __bf16 v16bf;
typedef __attribute__((ext_vector_type(8)))  __bf16 v8bf;
typedef __attribute__((ext_vector_type(8)))  float  v8f;

__device__ __forceinline__ __bf16 to_bf16(float f) {
    unsigned int u = __float_as_uint(f);
    unsigned int r = (u + 0x7FFFu + ((u >> 16) & 1u)) >> 16;   // RNE
    unsigned short s = (unsigned short)r;
    return __builtin_bit_cast(__bf16, s);
}

__device__ __forceinline__ v16bf cat16(v8bf a, v8bf b) {
    return __builtin_shufflevector(a, b, 0,1,2,3,4,5,6,7,8,9,10,11,12,13,14,15);
}

// 16 contiguous f32 (LDS or global) -> v16bf
__device__ __forceinline__ v16bf cvt16(const float* s) {
    float4 a = *(const float4*)(s);
    float4 b = *(const float4*)(s + 4);
    float4 c = *(const float4*)(s + 8);
    float4 d = *(const float4*)(s + 12);
    v16bf r;
    r[0]  = to_bf16(a.x); r[1]  = to_bf16(a.y); r[2]  = to_bf16(a.z); r[3]  = to_bf16(a.w);
    r[4]  = to_bf16(b.x); r[5]  = to_bf16(b.y); r[6]  = to_bf16(b.z); r[7]  = to_bf16(b.w);
    r[8]  = to_bf16(c.x); r[9]  = to_bf16(c.y); r[10] = to_bf16(c.z); r[11] = to_bf16(c.w);
    r[12] = to_bf16(d.x); r[13] = to_bf16(d.y); r[14] = to_bf16(d.z); r[15] = to_bf16(d.w);
    return r;
}

__device__ __forceinline__ v8f wmma_bf16(v16bf a, v16bf b, v8f c) {
    // v_wmma_f32_16x16x32_bf16 : D = A(16x32) * B(32x16) + C(16x16 f32)
    return __builtin_amdgcn_wmma_f32_16x16x32_bf16(false, a, false, b, (short)0, c, false, false);
}

__device__ __forceinline__ float rmax16(float x) {
    #pragma unroll
    for (int mm = 1; mm < 16; mm <<= 1) x = fmaxf(x, __shfl_xor(x, mm, 32));
    return x;
}
__device__ __forceinline__ float rsum16(float x) {
    #pragma unroll
    for (int mm = 1; mm < 16; mm <<= 1) x += __shfl_xor(x, mm, 32);
    return x;
}

// A-matrix fragments for a 16-row Q tile: lane holds row m=lane&15, half h=lane>>4.
// A 16x32 bf16 layout: element e -> K = 16*h + e, i.e. two runs of
// 8 contiguous d-values at d = 16h + 8p (+32 per kof).
__device__ __forceinline__ void load_q_frags(const float* __restrict__ qrow, int h, v16bf qa[2]) {
    #pragma unroll
    for (int kof = 0; kof < 2; ++kof) {
        #pragma unroll
        for (int p = 0; p < 2; ++p) {
            const float* src = qrow + kof * 32 + p * 8 + 16 * h;
            float4 f0 = *(const float4*)(src);
            float4 f1 = *(const float4*)(src + 4);
            const int e = p * 8;
            qa[kof][e + 0] = to_bf16(f0.x);
            qa[kof][e + 1] = to_bf16(f0.y);
            qa[kof][e + 2] = to_bf16(f0.z);
            qa[kof][e + 3] = to_bf16(f0.w);
            qa[kof][e + 4] = to_bf16(f1.x);
            qa[kof][e + 5] = to_bf16(f1.y);
            qa[kof][e + 6] = to_bf16(f1.z);
            qa[kof][e + 7] = to_bf16(f1.w);
        }
    }
}

// ---------------------------------------------------------------- zero bins
__global__ void psi_zero_bins(unsigned int* __restrict__ bins) {
    bins[blockIdx.x * 256 + threadIdx.x] = 0u;
}

// ---------------------------------------------------------------- histogram
__global__ __launch_bounds__(256) void psi_hist(const float* __restrict__ q,
                                                const float* __restrict__ k,
                                                unsigned int* __restrict__ gbins) {
    __shared__ __attribute__((aligned(16))) __bf16 ldsK[32 * DH];
    __shared__ unsigned int shist[NBINS];

    const int wave = threadIdx.x >> 5, lane = threadIdx.x & 31;
    const int m = lane & 15, h = lane >> 4;
    const int bh = blockIdx.x >> 3;
    const int q0 = (blockIdx.x & 7) * 128 + wave * 16;
    const float* qb = q + (size_t)bh * SEQ * DH;
    const float* kb = k + (size_t)bh * SEQ * DH;

    for (int i = threadIdx.x; i < NBINS; i += 256) shist[i] = 0u;

    v16bf qa[2];
    load_q_frags(qb + (size_t)(q0 + m) * DH, h, qa);

    for (int kblk = 0; kblk < SEQ; kblk += 32) {
        __syncthreads();
        {   // cooperative fill: K tile [32 keys][64 d] -> bf16 LDS
            const int i0 = threadIdx.x * 8;
            const float4* ks = (const float4*)(kb + (size_t)kblk * DH + i0);
            float4 a = ks[0], b = ks[1];
            float vals[8] = {a.x, a.y, a.z, a.w, b.x, b.y, b.z, b.w};
            #pragma unroll
            for (int j = 0; j < 8; ++j) ldsK[i0 + j] = to_bf16(vals[j]);
        }
        __syncthreads();

        #pragma unroll
        for (int t = 0; t < 2; ++t) {   // two 16-key subtiles
            const __bf16* kr = &ldsK[(t * 16 + m) * DH];
            v16bf bk0 = cat16(*(const v8bf*)(kr + 16 * h),      *(const v8bf*)(kr + 16 * h + 8));
            v16bf bk1 = cat16(*(const v8bf*)(kr + 32 + 16 * h), *(const v8bf*)(kr + 32 + 16 * h + 8));
            v8f c = {};
            c = wmma_bf16(qa[0], bk0, c);
            c = wmma_bf16(qa[1], bk1, c);
            #pragma unroll
            for (int r = 0; r < 8; ++r) {
                float s = c[r] * QK_SCALE;
                int bin = (int)floorf((s - HLO) * (1.0f / HBW));
                bin = bin < 0 ? 0 : (bin > NBINS - 1 ? NBINS - 1 : bin);
                atomicAdd(&shist[bin], 1u);     // ds_add_u32
            }
        }
    }
    __syncthreads();
    for (int i = threadIdx.x; i < NBINS; i += 256)
        if (shist[i]) atomicAdd(&gbins[i], shist[i]);
}

// ---------------------------------------------------------------- threshold
__global__ void psi_thresh(const unsigned int* __restrict__ bins, float* __restrict__ thr) {
    if (threadIdx.x == 0) {
        const double target = 0.5 * (double)(67108864ull - 1ull);   // quantile rank
        unsigned long long cum = 0;
        float result = 0.0f;
        for (int b = 0; b < NBINS; ++b) {
            unsigned int c = bins[b];
            if ((double)cum + (double)c > target) {
                double frac = (target - (double)cum) / (double)(c ? c : 1u);
                result = HLO + ((float)b + (float)frac) * HBW;
                break;
            }
            cum += c;
        }
        *thr = result;
    }
}

// ---------------------------------------------------------------- flash pass
// K/V^T tiles staged f32 in LDS by global_load_async_to_lds_* (ASYNCcnt),
// double-buffered so tile i+1 streams in while tile i runs WMMA + softmax.
__global__ __launch_bounds__(256) void psi_attn(const float* __restrict__ q,
                                                const float* __restrict__ k,
                                                const float* __restrict__ v,
                                                const float* __restrict__ thrp,
                                                float* __restrict__ out) {
    __shared__ __attribute__((aligned(16))) float  ldsKf[2][32 * DH];   // [key][d]
    __shared__ __attribute__((aligned(16))) float  ldsVT[2][DH * 32];   // [d][key]
    __shared__ __attribute__((aligned(16))) __bf16 ldsP[8][16 * 32];    // per-wave P tile

    const int wave = threadIdx.x >> 5, lane = threadIdx.x & 31;
    const int m = lane & 15, h = lane >> 4;
    const int bh = blockIdx.x >> 3;
    const int q0 = (blockIdx.x & 7) * 128 + wave * 16;
    const float* qb = q + (size_t)bh * SEQ * DH;
    const float* kb = k + (size_t)bh * SEQ * DH;
    const float* vb = v + (size_t)bh * SEQ * DH;
    float* ob = out + (size_t)bh * SEQ * DH;
    const float thr = *thrp;

    const int i0 = threadIdx.x * 8;   // this thread's 8 elements of the 32x64 tile

    // Async fill of buffer `buf` with key-block starting at `blk`.
    // K: 2x b128 (32B contiguous per lane, instruction offset applies to both
    // global and LDS sides). V: 8x b32 scattered to the transposed LDS layout.
    auto issue_fill = [&](int blk, int buf) {
        const float* gk = kb + (size_t)blk * DH + i0;
        unsigned kdst = (unsigned)(size_t)&ldsKf[buf][i0];
        asm volatile("global_load_async_to_lds_b128 %0, %1, off"
                     :: "v"(kdst), "v"(gk) : "memory");
        asm volatile("global_load_async_to_lds_b128 %0, %1, off offset:16"
                     :: "v"(kdst), "v"(gk) : "memory");
        #pragma unroll
        for (int j = 0; j < 8; ++j) {
            const int idx = i0 + j, key = idx >> 6, d = idx & 63;
            const float* gv = vb + (size_t)blk * DH + idx;
            unsigned vdst = (unsigned)(size_t)&ldsVT[buf][d * 32 + key];
            asm volatile("global_load_async_to_lds_b32 %0, %1, off"
                         :: "v"(vdst), "v"(gv) : "memory");
        }
    };

    v16bf qa[2];
    load_q_frags(qb + (size_t)(q0 + m) * DH, h, qa);

    v8f acc[4] = {{}, {}, {}, {}};
    float mrun[8], lrun[8];
    #pragma unroll
    for (int r = 0; r < 8; ++r) { mrun[r] = NEGMAX; lrun[r] = 0.0f; }

    issue_fill(0, 0);
    asm volatile("s_wait_asynccnt 0x0" ::: "memory");
    __syncthreads();

    for (int kblk = 0; kblk < SEQ; kblk += 32) {
        const int buf = (kblk >> 5) & 1;
        if (kblk + 32 < SEQ) issue_fill(kblk + 32, buf ^ 1);   // overlap with compute

        // QK^T: two 16x16 score tiles (keys kblk..+15, kblk+16..+31).
        // B 32x16 layout: lane -> N = key = t*16+m, element e -> K(d) = kof*32+16h+e.
        v8f ct[2];
        #pragma unroll
        for (int t = 0; t < 2; ++t) {
            const float* krf = &ldsKf[buf][(t * 16 + m) * DH + 16 * h];
            v16bf bk0 = cvt16(krf);
            v16bf bk1 = cvt16(krf + 32);
            v8f c = {};
            c = wmma_bf16(qa[0], bk0, c);
            c = wmma_bf16(qa[1], bk1, c);
            ct[t] = c;
        }

        // scale + global-threshold mask + online softmax (rows r+8h, N across 16 lanes)
        float s0[8], s1[8], p0a[8], p1a[8];
        #pragma unroll
        for (int r = 0; r < 8; ++r) {
            float a0 = ct[0][r] * QK_SCALE; s0[r] = (a0 > thr) ? a0 : NEGMAX;
            float a1 = ct[1][r] * QK_SCALE; s1[r] = (a1 > thr) ? a1 : NEGMAX;
        }
        #pragma unroll
        for (int r = 0; r < 8; ++r) {
            float tmax = rmax16(fmaxf(s0[r], s1[r]));
            float mnew = fmaxf(mrun[r], tmax);
            float corr = __expf(mrun[r] - mnew);            // finite: NEGMAX, not -inf
            float p0 = (s0[r] > -3.3e38f) ? __expf(s0[r] - mnew) : 0.0f;
            float p1 = (s1[r] > -3.3e38f) ? __expf(s1[r] - mnew) : 0.0f;
            float tsum = rsum16(p0 + p1);
            lrun[r] = lrun[r] * corr + tsum;
            mrun[r] = mnew;
            #pragma unroll
            for (int db = 0; db < 4; ++db) acc[db][r] *= corr;
            p0a[r] = p0; p1a[r] = p1;
        }

        // C-layout -> A-layout for P via per-wave LDS roundtrip (bf16)
        __bf16* prow = &ldsP[wave][0];
        #pragma unroll
        for (int r = 0; r < 8; ++r) {
            int row = r + 8 * h;
            prow[row * 32 + m]      = to_bf16(p0a[r]);
            prow[row * 32 + 16 + m] = to_bf16(p1a[r]);
        }
        asm volatile("s_wait_dscnt 0" ::: "memory");
        const __bf16* pm = &prow[m * 32];
        v16bf pa = cat16(*(const v8bf*)(pm + 16 * h), *(const v8bf*)(pm + 16 * h + 8));

        // PV: A = P(16x32 keys), B = V^T slices (32 keys x 16 d-cols)
        #pragma unroll
        for (int db = 0; db < 4; ++db) {
            const float* vrf = &ldsVT[buf][(db * 16 + m) * 32 + 16 * h];
            v16bf vbf = cvt16(vrf);
            acc[db] = wmma_bf16(pa, vbf, acc[db]);
        }

        // next tile's async fills must land before the next iteration reads them
        asm volatile("s_wait_asynccnt 0x0" ::: "memory");
        __syncthreads();
    }

    // epilogue: normalize, scatter C layout to global
    #pragma unroll
    for (int r = 0; r < 8; ++r) {
        float inv = lrun[r] > 0.0f ? 1.0f / lrun[r] : 0.0f;
        int row = q0 + r + 8 * h;
        #pragma unroll
        for (int db = 0; db < 4; ++db)
            ob[(size_t)row * DH + db * 16 + m] = acc[db][r] * inv;
    }
}

extern "C" void kernel_launch(void* const* d_in, const int* in_sizes, int n_in,
                              void* d_out, int out_size, void* d_ws, size_t ws_size,
                              hipStream_t stream) {
    const float* q = (const float*)d_in[0];
    const float* k = (const float*)d_in[1];
    const float* v = (const float*)d_in[2];
    float* out = (float*)d_out;

    float* thr = (float*)d_ws;
    unsigned int* bins = (unsigned int*)((char*)d_ws + 256);

    psi_zero_bins<<<NBINS / 256, 256, 0, stream>>>(bins);
    psi_hist<<<BH * (SEQ / 128), 256, 0, stream>>>(q, k, bins);
    psi_thresh<<<1, 64, 0, stream>>>(bins, thr);
    psi_attn<<<BH * (SEQ / 128), 256, 0, stream>>>(q, k, v, thr, out);
}
